// VQVAE_2851858284843
// MI455X (gfx1250) — matrix operand order repaired and checked by reading, not compiled, not run on hardware
//
#include <hip/hip_runtime.h>
#include <hip/hip_bf16.h>

// ---------------------------------------------------------------------------
// Types for CDNA5 WMMA (wave32)
// ---------------------------------------------------------------------------
typedef __attribute__((ext_vector_type(16))) __bf16 v16bf;
typedef __attribute__((ext_vector_type(8)))  __bf16 v8bf;
typedef __attribute__((ext_vector_type(8)))  float  v8f;

union AF { v16bf v; v8bf h[2]; };

// Problem constants
#define BATCH 32768
#define ADIM  64
#define HDIM  1024
#define DDIM  256
#define KCODES 4096

// ---------------------------------------------------------------------------
// CDNA5 async global->LDS DMA (ASYNCcnt-tracked), inline asm (no builtin).
// lds_off: byte offset within the wave's LDS (low 32 bits of a generic
// pointer to a __shared__ object). gaddr: 64-bit global address.
// ---------------------------------------------------------------------------
__device__ __forceinline__ void async_ld16(unsigned lds_off, uint64_t gaddr) {
  asm volatile("global_load_async_to_lds_b128 %0, %1, off"
               :: "v"(lds_off), "v"(gaddr) : "memory");
}
__device__ __forceinline__ void wait_async0() {
  asm volatile("s_wait_asynccnt 0" ::: "memory");
}
__device__ __forceinline__ unsigned lds_off32(const void* p) {
  return (unsigned)(uintptr_t)p;   // low 32 bits of generic ptr = LDS offset
}

// ---------------------------------------------------------------------------
// Small helper kernels
// ---------------------------------------------------------------------------
__global__ void conv_bf16_kernel(const float* __restrict__ in,
                                 __bf16* __restrict__ out, int n) {
  int i = blockIdx.x * 256 + threadIdx.x;
  if (i < n) out[i] = (__bf16)in[i];
}

// in: f32 [Kd][Nd] row-major -> out: bf16 [Nd][Kd] row-major (transposed)
__global__ void transpose_bf16_kernel(const float* __restrict__ in,
                                      __bf16* __restrict__ out, int Kd, int Nd) {
  int i = blockIdx.x * 256 + threadIdx.x;
  if (i >= Kd * Nd) return;
  int k = i / Nd, n = i % Nd;
  out[(size_t)n * Kd + k] = (__bf16)in[i];
}

// squared L2 norm per codebook row: cn[k] = sum_d cb[k][d]^2
__global__ void cnorm_kernel(const float* __restrict__ cb, float* __restrict__ cn) {
  int k = blockIdx.x;               // KCODES blocks, 256 threads
  float v = cb[(size_t)k * DDIM + threadIdx.x];
  float s = v * v;
  for (int o = 16; o; o >>= 1) s += __shfl_down(s, o);
  __shared__ float w[8];
  if ((threadIdx.x & 31) == 0) w[threadIdx.x >> 5] = s;
  __syncthreads();
  if (threadIdx.x == 0) {
    float t = 0.f;
    for (int i = 0; i < 8; ++i) t += w[i];
    cn[k] = t;
  }
}

// ---------------------------------------------------------------------------
// Generic bf16 WMMA GEMM:  out = epilogue(A[M,K] @ Bt[N,K]^T + bias)
// BM=128, BK=32, 256 threads = 8 waves. Double-buffered LDS tiles filled by
// GLOBAL_LOAD_ASYNC_TO_LDS_B128; one barrier per k-step; DMA for tile t+1
// overlaps WMMA of tile t.
//   EPI 0: relu -> bf16 out
//   EPI 1: bias only -> f32 out AND bf16 out (for mu/enc)
//   EPI 2: tanh, then sum (tanh(x)-ref)^2 into *sum_out (atomic)
// ---------------------------------------------------------------------------
#define BK 32
#define LDT 40   // LDS row stride in bf16 elems (32 + 8 pad, keeps 16B align)

template <int BN, int NFM, int NFN, int EPI>
__global__ __launch_bounds__(256)
void gemm_kernel(const __bf16* __restrict__ Ag, const __bf16* __restrict__ Btg,
                 const float* __restrict__ bias,
                 __bf16* __restrict__ outb, float* __restrict__ outf,
                 const float* __restrict__ ref, float* __restrict__ sum_out,
                 int M, int N, int K) {
  constexpr int BMc = 128;
  constexpr int WAVES_N = BN / (NFN * 16);
  constexpr int WAVES_M = BMc / (NFM * 16);
  static_assert(WAVES_M * WAVES_N == 8, "need 8 waves");

  __shared__ __align__(16) __bf16 lsA[2][BMc * LDT];
  __shared__ __align__(16) __bf16 lsB[2][BN * LDT];

  const int t    = threadIdx.x;
  const int wid  = t >> 5;
  const int lane = t & 31;
  const int l16  = lane & 15;
  const int hi   = lane >> 4;           // 0: lanes 0-15, 1: lanes 16-31
  const int wm   = wid % WAVES_M;
  const int wn   = wid / WAVES_M;
  const int m0   = blockIdx.x * BMc;
  const int n0   = blockIdx.y * BN;
  const int aRow0 = wm * NFM * 16;
  const int bCol0 = wn * NFN * 16;

  auto issue_tile = [&](int k0, int buf) {
    #pragma unroll
    for (int c = t; c < BMc * 4; c += 256) {
      int row = c >> 2, cc = c & 3;
      async_ld16(lds_off32(&lsA[buf][row * LDT + cc * 8]),
                 (uint64_t)(uintptr_t)&Ag[(size_t)(m0 + row) * K + k0 + cc * 8]);
    }
    #pragma unroll
    for (int c = t; c < BN * 4; c += 256) {
      int row = c >> 2, cc = c & 3;
      async_ld16(lds_off32(&lsB[buf][row * LDT + cc * 8]),
                 (uint64_t)(uintptr_t)&Btg[(size_t)(n0 + row) * K + k0 + cc * 8]);
    }
  };

  v8f acc[NFM][NFN] = {};
  const int nk = K / BK;
  issue_tile(0, 0);

  for (int kt = 0; kt < nk; ++kt) {
    const int cur = kt & 1;
    wait_async0();          // tile kt landed in LDS (this wave's DMAs)
    __syncthreads();        // all waves' DMAs visible; buf cur^1 free to reuse
    if (kt + 1 < nk) issue_tile((kt + 1) * BK, cur ^ 1);  // overlaps compute

    AF a[NFM], b[NFN];
    #pragma unroll
    for (int mi = 0; mi < NFM; ++mi) {
      int row = aRow0 + mi * 16 + l16;
      int ak  = hi * 8;                 // ISA A layout: lo lanes K0..7/16..23
      a[mi].h[0] = *(const v8bf*)&lsA[cur][row * LDT + ak];
      a[mi].h[1] = *(const v8bf*)&lsA[cur][row * LDT + ak + 16];
    }
    #pragma unroll
    for (int ni = 0; ni < NFN; ++ni) {
      int row = bCol0 + ni * 16 + l16;
      int bk  = hi * 8;                 // B layout: lo lanes K0..7/16..23, hi K8..15/24..31
      b[ni].h[0] = *(const v8bf*)&lsB[cur][row * LDT + bk];
      b[ni].h[1] = *(const v8bf*)&lsB[cur][row * LDT + bk + 16];
    }
    #pragma unroll
    for (int mi = 0; mi < NFM; ++mi)
      #pragma unroll
      for (int ni = 0; ni < NFN; ++ni)
        acc[mi][ni] = __builtin_amdgcn_wmma_f32_16x16x32_bf16(
            false, a[mi].v, false, b[ni].v, (short)0, acc[mi][ni], false, false);
  }

  // ---- epilogue (C layout: lanes 0-15 rows r, lanes 16-31 rows 8+r) ----
  float lsum = 0.f;
  #pragma unroll
  for (int ni = 0; ni < NFN; ++ni) {
    int col = n0 + bCol0 + ni * 16 + l16;
    float bcol = bias[col];
    #pragma unroll
    for (int mi = 0; mi < NFM; ++mi) {
      #pragma unroll
      for (int r = 0; r < 8; ++r) {
        int row = m0 + aRow0 + mi * 16 + hi * 8 + r;
        float v = acc[mi][ni][r] + bcol;
        size_t o = (size_t)row * N + col;
        if constexpr (EPI == 0) {
          outb[o] = (__bf16)fmaxf(v, 0.f);
        } else if constexpr (EPI == 1) {
          outf[o] = v;
          outb[o] = (__bf16)v;
        } else {
          float tv = tanhf(v);
          float d  = tv - ref[o];
          lsum += d * d;
        }
      }
    }
  }
  if constexpr (EPI == 2) {
    for (int off = 16; off; off >>= 1) lsum += __shfl_down(lsum, off);
    if (lane == 0) atomicAdd(sum_out, lsum);
  }
}

// ---------------------------------------------------------------------------
// Fused VQ distance + argmin kernel.
// 128 enc rows staged once in LDS (async DMA); 64-code tiles of the codebook
// double-buffered with async DMA overlapping the WMMA of the previous tile.
// score = cnorm[k] - 2*e.c  (||e||^2 irrelevant for argmin); running
// (min, argmin) per row in registers, butterfly-reduced across the 16-lane
// half-groups matching the WMMA C layout.
// ---------------------------------------------------------------------------
#define VQ_LD 264   // 256 + 8 pad, 16B aligned stride

__global__ __launch_bounds__(256)
void vq_argmin_kernel(const __bf16* __restrict__ enc_bf,
                      const __bf16* __restrict__ cb_bf,
                      const float* __restrict__ cnorm,
                      int* __restrict__ idx_out) {
  extern __shared__ __align__(16) __bf16 lds[];
  __bf16* la = lds;                          // 128 * VQ_LD
  __bf16* lb0 = lds + 128 * VQ_LD;           // 64 * VQ_LD
  __bf16* lb1 = lb0 + 64 * VQ_LD;            // 64 * VQ_LD

  const int t    = threadIdx.x;
  const int wid  = t >> 5;
  const int lane = t & 31;
  const int l16  = lane & 15;
  const int hi   = lane >> 4;
  const int m0   = blockIdx.x * 128;

  auto issue_codes = [&](int nt, __bf16* dst) {
    #pragma unroll
    for (int c = t; c < 64 * 32; c += 256) {
      int row = c >> 5, cc = c & 31;
      async_ld16(lds_off32(&dst[row * VQ_LD + cc * 8]),
                 (uint64_t)(uintptr_t)&cb_bf[(size_t)(nt * 64 + row) * DDIM + cc * 8]);
    }
  };

  // stage 128 enc rows x 256 (bf16) once + first code tile
  #pragma unroll
  for (int c = t; c < 128 * 32; c += 256) {
    int row = c >> 5, cc = c & 31;
    async_ld16(lds_off32(&la[row * VQ_LD + cc * 8]),
               (uint64_t)(uintptr_t)&enc_bf[(size_t)(m0 + row) * DDIM + cc * 8]);
  }
  issue_codes(0, lb0);

  float rmin[8];
  int   ridx[8];
  #pragma unroll
  for (int r = 0; r < 8; ++r) { rmin[r] = 3.4e38f; ridx[r] = 0; }

  constexpr int NT = KCODES / 64;
  for (int nt = 0; nt < NT; ++nt) {
    __bf16* cb_cur = (nt & 1) ? lb1 : lb0;
    __bf16* cb_nxt = (nt & 1) ? lb0 : lb1;
    wait_async0();
    __syncthreads();
    if (nt + 1 < NT) issue_codes(nt + 1, cb_nxt);   // overlaps WMMA below

    v8f acc[4] = {};
    #pragma unroll
    for (int ks = 0; ks < DDIM / 32; ++ks) {
      AF a;
      int arow = wid * 16 + l16;
      int ak   = ks * 32 + hi * 8;
      a.h[0] = *(const v8bf*)&la[arow * VQ_LD + ak];
      a.h[1] = *(const v8bf*)&la[arow * VQ_LD + ak + 16];
      AF b[4];
      #pragma unroll
      for (int ni = 0; ni < 4; ++ni) {
        int brow = ni * 16 + l16;
        int bk   = ks * 32 + hi * 8;
        b[ni].h[0] = *(const v8bf*)&cb_cur[brow * VQ_LD + bk];
        b[ni].h[1] = *(const v8bf*)&cb_cur[brow * VQ_LD + bk + 16];
      }
      #pragma unroll
      for (int ni = 0; ni < 4; ++ni)
        acc[ni] = __builtin_amdgcn_wmma_f32_16x16x32_bf16(
            false, a.v, false, b[ni].v, (short)0, acc[ni], false, false);
    }

    float cn[4];
    #pragma unroll
    for (int ni = 0; ni < 4; ++ni) cn[ni] = cnorm[nt * 64 + ni * 16 + l16];

    #pragma unroll
    for (int r = 0; r < 8; ++r) {
      float v = cn[0] - 2.0f * acc[0][r];
      int  ix = nt * 64 + l16;
      #pragma unroll
      for (int ni = 1; ni < 4; ++ni) {
        float v2 = cn[ni] - 2.0f * acc[ni][r];
        int  ix2 = nt * 64 + ni * 16 + l16;
        if (v2 < v) { v = v2; ix = ix2; }
      }
      // butterfly argmin across the 16-lane half (stays within half-group)
      #pragma unroll
      for (int o = 1; o < 16; o <<= 1) {
        float ov = __shfl_xor(v, o);
        int   oi = __shfl_xor(ix, o);
        if (ov < v || (ov == v && oi < ix)) { v = ov; ix = oi; }
      }
      if (v < rmin[r]) { rmin[r] = v; ridx[r] = ix; }
    }
  }

  if (l16 == 0) {
    #pragma unroll
    for (int r = 0; r < 8; ++r) {
      int row = m0 + wid * 16 + hi * 8 + r;
      idx_out[row] = ridx[r];
    }
  }
}

// ---------------------------------------------------------------------------
// Gather q = codebook[idx], emit q in bf16 for decoder, accumulate
// sum (q - enc)^2  (== both embedding and commitment loss numerators).
// ---------------------------------------------------------------------------
__global__ void gather_loss_kernel(const int* __restrict__ idx,
                                   const float* __restrict__ codebook,
                                   const float* __restrict__ enc_f,
                                   __bf16* __restrict__ q_bf,
                                   float* __restrict__ sum_vq) {
  int row = blockIdx.x;     // BATCH blocks, 256 threads = DDIM
  int d   = threadIdx.x;
  int id  = idx[row];
  float c = codebook[(size_t)id * DDIM + d];
  float e = enc_f[(size_t)row * DDIM + d];
  q_bf[(size_t)row * DDIM + d] = (__bf16)c;
  float diff = c - e;
  float s = diff * diff;
  for (int o = 16; o; o >>= 1) s += __shfl_down(s, o);
  __shared__ float w[8];
  if ((threadIdx.x & 31) == 0) w[threadIdx.x >> 5] = s;
  __syncthreads();
  if (threadIdx.x == 0) {
    float tot = 0.f;
    for (int i = 0; i < 8; ++i) tot += w[i];
    atomicAdd(sum_vq, tot);
  }
}

// ---------------------------------------------------------------------------
// Final scalar assembly:
//   mse_vq = sum_vq / (B*D);  rec = sum_rec / (B*A)
//   vq_loss = BETA*c + e = 1.25 * mse_vq; total = rec + vq_loss
// ---------------------------------------------------------------------------
__global__ void finalize_kernel(const float* __restrict__ sums,
                                float* __restrict__ out) {
  float mse_vq = sums[0] / (float)((size_t)BATCH * DDIM);
  float rec    = sums[1] / (float)((size_t)BATCH * ADIM);
  float vq     = 1.25f * mse_vq;
  out[0] = rec + vq;   // total
  out[1] = rec;        // recons_loss
  out[2] = vq;         // vq_loss
  out[3] = mse_vq;     // embedding_loss
  out[4] = mse_vq;     // commitment_loss
}

// ---------------------------------------------------------------------------
// Host launcher
// ---------------------------------------------------------------------------
extern "C" void kernel_launch(void* const* d_in, const int* in_sizes, int n_in,
                              void* d_out, int out_size, void* d_ws, size_t ws_size,
                              hipStream_t stream) {
  const float* action = (const float*)d_in[0];
  const float* enc_w1 = (const float*)d_in[1];
  const float* enc_b1 = (const float*)d_in[2];
  const float* enc_w2 = (const float*)d_in[3];
  const float* enc_b2 = (const float*)d_in[4];
  const float* mu_w   = (const float*)d_in[5];
  const float* mu_b   = (const float*)d_in[6];
  const float* codebook = (const float*)d_in[7];
  const float* dec_w1 = (const float*)d_in[8];
  const float* dec_b1 = (const float*)d_in[9];
  const float* dec_w2 = (const float*)d_in[10];
  const float* dec_b2 = (const float*)d_in[11];
  const float* dec_w3 = (const float*)d_in[12];
  const float* dec_b3 = (const float*)d_in[13];
  float* out = (float*)d_out;

  // workspace carve-up (256B aligned)
  char* base = (char*)d_ws;
  size_t off = 0;
  auto alloc = [&](size_t bytes) -> char* {
    char* p = base + off;
    off = (off + bytes + 255) & ~(size_t)255;
    return p;
  };
  float*  sums   = (float*)alloc(2 * sizeof(float));                 // [vq, rec]
  __bf16* act_bf = (__bf16*)alloc((size_t)BATCH * ADIM * 2);
  __bf16* w1t    = (__bf16*)alloc((size_t)HDIM * ADIM * 2);          // [H][A]
  __bf16* w2t    = (__bf16*)alloc((size_t)HDIM * HDIM * 2);          // [H][H]
  __bf16* muwt   = (__bf16*)alloc((size_t)DDIM * HDIM * 2);          // [D][H]
  __bf16* d1t    = (__bf16*)alloc((size_t)HDIM * DDIM * 2);          // [H][D]
  __bf16* d2t    = (__bf16*)alloc((size_t)HDIM * HDIM * 2);          // [H][H]
  __bf16* d3t    = (__bf16*)alloc((size_t)ADIM * HDIM * 2);          // [A][H]
  __bf16* cb_bf  = (__bf16*)alloc((size_t)KCODES * DDIM * 2);
  float*  cnorm  = (float*)alloc((size_t)KCODES * 4);
  __bf16* h1     = (__bf16*)alloc((size_t)BATCH * HDIM * 2);         // reused as g1
  __bf16* h2     = (__bf16*)alloc((size_t)BATCH * HDIM * 2);         // reused as g2
  float*  enc_f  = (float*)alloc((size_t)BATCH * DDIM * 4);
  __bf16* enc_bf = (__bf16*)alloc((size_t)BATCH * DDIM * 2);
  int*    idxbuf = (int*)alloc((size_t)BATCH * 4);
  __bf16* q_bf   = (__bf16*)alloc((size_t)BATCH * DDIM * 2);
  (void)ws_size; (void)n_in; (void)in_sizes; (void)out_size;

  hipMemsetAsync(sums, 0, 2 * sizeof(float), stream);

  // converts / transposes (weights -> bf16 [N][K])
  auto cdiv = [](int a, int b) { return (a + b - 1) / b; };
  conv_bf16_kernel<<<cdiv(BATCH * ADIM, 256), 256, 0, stream>>>(action, act_bf, BATCH * ADIM);
  transpose_bf16_kernel<<<cdiv(ADIM * HDIM, 256), 256, 0, stream>>>(enc_w1, w1t, ADIM, HDIM);
  transpose_bf16_kernel<<<cdiv(HDIM * HDIM, 256), 256, 0, stream>>>(enc_w2, w2t, HDIM, HDIM);
  transpose_bf16_kernel<<<cdiv(HDIM * DDIM, 256), 256, 0, stream>>>(mu_w, muwt, HDIM, DDIM);
  transpose_bf16_kernel<<<cdiv(DDIM * HDIM, 256), 256, 0, stream>>>(dec_w1, d1t, DDIM, HDIM);
  transpose_bf16_kernel<<<cdiv(HDIM * HDIM, 256), 256, 0, stream>>>(dec_w2, d2t, HDIM, HDIM);
  transpose_bf16_kernel<<<cdiv(HDIM * ADIM, 256), 256, 0, stream>>>(dec_w3, d3t, HDIM, ADIM);
  conv_bf16_kernel<<<cdiv(KCODES * DDIM, 256), 256, 0, stream>>>(codebook, cb_bf, KCODES * DDIM);
  cnorm_kernel<<<KCODES, 256, 0, stream>>>(codebook, cnorm);

  // encoder
  gemm_kernel<128, 2, 4, 0><<<dim3(BATCH / 128, HDIM / 128), 256, 0, stream>>>(
      act_bf, w1t, enc_b1, h1, nullptr, nullptr, nullptr, BATCH, HDIM, ADIM);
  gemm_kernel<128, 2, 4, 0><<<dim3(BATCH / 128, HDIM / 128), 256, 0, stream>>>(
      h1, w2t, enc_b2, h2, nullptr, nullptr, nullptr, BATCH, HDIM, HDIM);
  gemm_kernel<128, 2, 4, 1><<<dim3(BATCH / 128, DDIM / 128), 256, 0, stream>>>(
      h2, muwt, mu_b, enc_bf, enc_f, nullptr, nullptr, BATCH, DDIM, HDIM);

  // VQ: fused distance GEMM + argmin, then gather + vq loss
  size_t vq_lds = (size_t)(128 * VQ_LD + 2 * 64 * VQ_LD) * sizeof(__bf16);
  vq_argmin_kernel<<<BATCH / 128, 256, vq_lds, stream>>>(enc_bf, cb_bf, cnorm, idxbuf);
  gather_loss_kernel<<<BATCH, 256, 0, stream>>>(idxbuf, codebook, enc_f, q_bf, &sums[0]);

  // decoder (q_st == q numerically); reuse h1/h2 buffers
  gemm_kernel<128, 2, 4, 0><<<dim3(BATCH / 128, HDIM / 128), 256, 0, stream>>>(
      q_bf, d1t, dec_b1, h1, nullptr, nullptr, nullptr, BATCH, HDIM, DDIM);
  gemm_kernel<128, 2, 4, 0><<<dim3(BATCH / 128, HDIM / 128), 256, 0, stream>>>(
      h1, d2t, dec_b2, h2, nullptr, nullptr, nullptr, BATCH, HDIM, HDIM);
  gemm_kernel<64, 1, 4, 2><<<dim3(BATCH / 128, 1), 256, 0, stream>>>(
      h2, d3t, dec_b3, nullptr, nullptr, action, &sums[1], BATCH, ADIM, HDIM);

  finalize_kernel<<<1, 1, 0, stream>>>(sums, out);
}
